// CSCOut_77781857730663
// MI455X (gfx1250) — matrix operand order repaired and checked by reading, not compiled, not run on hardware
//
#include <hip/hip_runtime.h>
#include <hip/hip_bf16.h>
#include <math.h>

// ---------------------------------------------------------------------------
// MI455X fused equivariant tensor product.
//   - wpost folded into gw2 => 2nd GEMM is 64->144 instead of 64->2304
//   - both GEMMs via v_wmma_f32_16x16x32_f16 (wave32, 16x16 tiles)
//   - per-lane tail phase uses packed float2 math (targets v_pk_fma_f32)
//   - x_spherical row prefetched at kernel entry (global_prefetch_b8)
//   - CG tables / QF computed once on device (prep kernel, double precision)
//   - kernel is memory-bound (~80MB @ 23.3TB/s ~ 3.5us floor)
// ---------------------------------------------------------------------------

typedef __attribute__((ext_vector_type(16))) _Float16 v16h;
typedef __attribute__((ext_vector_type(8)))  float    v8f;
typedef __attribute__((ext_vector_type(2)))  _Float16 h2v;
typedef __attribute__((ext_vector_type(2)))  float    f2;

#define N_ATOM 32768

// instruction table: (i, j, lo), position within lo-group, group size
__device__ __constant__ const int INS_I[9]   = {0,0,1,1,1,2,2,2,2};
__device__ __constant__ const int INS_J[9]   = {0,2,1,1,1,0,2,2,2};
__device__ __constant__ const int INS_LO[9]  = {0,2,0,1,2,2,0,1,2};
__device__ __constant__ const int INS_POS[9] = {0,0,1,0,1,2,2,1,3};
__device__ __constant__ const int INS_NLO[9] = {3,4,3,2,4,4,3,2,4};
// CG storage offsets (floats): sizes 1,25,9,27,45,25,25,75,125 -> total 357
#define CG0 0
#define CG1 1
#define CG2 26
#define CG3 35
#define CG4 62
#define CG5 107
#define CG6 132
#define CG7 157
#define CG8 232

// ws byte offsets
#define WS_CG    0        // 357 f32
#define WS_QFP   1536     // 81 f32
#define WS_B2    1920     // 144 f32
#define WS_GW1T  2560     // 128x64 half, stored [n][k] (K-transposed), 16384 B
#define WS_A2T   18944    // 144x64 half, stored [q][h] (K-transposed), 18432 B

// ------------------------- prep: CG / QF / folded weights -------------------

struct DC { double r, i; };
__device__ inline DC cmul(DC a, DC b){ return DC{a.r*b.r - a.i*b.i, a.r*b.i + a.i*b.r}; }

__device__ double su2_cg_dev(int j1,int j2,int j3,int m1,int m2,int m3){
  const double F[8] = {1.,1.,2.,6.,24.,120.,720.,5040.};
  if (m3 != m1 + m2) return 0.0;
  int dj = j1 - j2; if (dj < 0) dj = -dj;
  if (j3 < dj || j3 > j1 + j2) return 0.0;
  double pref = sqrt((2.0*j3+1.0)*F[j3+j1-j2]*F[j3-j1+j2]*F[j1+j2-j3]/F[j1+j2+j3+1]);
  pref *= sqrt(F[j3+m3]*F[j3-m3]*F[j1-m1]*F[j1+m1]*F[j2-m2]*F[j2+m2]);
  int v0 = 0;
  if (j2 - j3 - m1 > v0) v0 = j2 - j3 - m1;
  if (j1 - j3 + m2 > v0) v0 = j1 - j3 + m2;
  int v1 = j1 + j2 - j3;
  if (j1 - m1 < v1) v1 = j1 - m1;
  if (j2 + m2 < v1) v1 = j2 + m2;
  double s = 0.0;
  for (int v = v0; v <= v1; ++v){
    double d = F[v]*F[j1+j2-j3-v]*F[j1-m1-v]*F[j2+m2-v]*F[j3-j2+m1+v]*F[j3-j1-m2+v];
    s += ((v & 1) ? -1.0 : 1.0) / d;
  }
  return pref * s;
}

__device__ void r2c_dev(int l, DC q[5][5]){
  for (int a=0;a<5;++a) for (int b=0;b<5;++b) q[a][b] = DC{0.0,0.0};
  double s = 1.0 / sqrt(2.0);
  for (int m=1;m<=l;++m){              // m negative
    q[l-m][l+m] = DC{s, 0.0};
    q[l-m][l-m] = DC{0.0, -s};
  }
  q[l][l] = DC{1.0, 0.0};
  for (int m=1;m<=l;++m){
    double sg = (m & 1) ? -1.0 : 1.0;
    q[l+m][l+m] = DC{sg*s, 0.0};
    q[l+m][l-m] = DC{0.0, sg*s};
  }
  DC ph;
  switch (l & 3){
    case 0: ph = DC{1.0,0.0};  break;
    case 1: ph = DC{0.0,-1.0}; break;
    case 2: ph = DC{-1.0,0.0}; break;
    default: ph = DC{0.0,1.0}; break;
  }
  for (int a=0;a<5;++a) for (int b=0;b<5;++b) q[a][b] = cmul(q[a][b], ph);
}

__device__ void real_cg_dev(int l1,int l2,int l3, float* out){
  int d1 = 2*l1+1, d2 = 2*l2+1, d3 = 2*l3+1;
  double C[5][5][5];
  for (int a=0;a<5;++a) for (int b=0;b<5;++b) for (int c=0;c<5;++c) C[a][b][c]=0.0;
  for (int m1=-l1;m1<=l1;++m1)
    for (int m2=-l2;m2<=l2;++m2){
      int m3 = m1 + m2;
      int am3 = m3 < 0 ? -m3 : m3;
      if (am3 <= l3) C[l1+m1][l2+m2][l3+m3] = su2_cg_dev(l1,l2,l3,m1,m2,m3);
    }
  DC q1[5][5], q2[5][5], q3[5][5];
  r2c_dev(l1,q1); r2c_dev(l2,q2); r2c_dev(l3,q3);
  double Rr[5][5][5], Ri[5][5][5];
  double nr = 0.0, ni = 0.0;
  for (int a=0;a<d1;++a)
    for (int b=0;b<d2;++b)
      for (int c=0;c<d3;++c){
        DC acc{0.0,0.0};
        for (int i=0;i<d1;++i)
          for (int k=0;k<d2;++k)
            for (int n=0;n<d3;++n){
              double cv = C[i][k][n];
              if (cv == 0.0) continue;
              DC q3c = DC{q3[n][c].r, -q3[n][c].i};       // conj
              DC t = cmul(cmul(q1[i][a], q2[k][b]), q3c);
              acc.r += t.r * cv; acc.i += t.i * cv;
            }
        Rr[a][b][c] = acc.r; Ri[a][b][c] = acc.i;
        nr += acc.r*acc.r; ni += acc.i*acc.i;
      }
  bool useRe = (sqrt(nr) >= sqrt(ni));
  double nn = useRe ? sqrt(nr) : sqrt(ni);
  double inv = (nn > 0.0) ? 1.0/nn : 0.0;
  for (int a=0;a<d1;++a)
    for (int b=0;b<d2;++b)
      for (int c=0;c<d3;++c)
        out[(a*d2+b)*d3+c] = (float)((useRe ? Rr[a][b][c] : Ri[a][b][c]) * inv);
}

__global__ __launch_bounds__(256)
void csc_prep_kernel(const float* __restrict__ gw1, const float* __restrict__ gw2,
                     const float* __restrict__ gb2,
                     const float* __restrict__ wpost0, const float* __restrict__ wpost1,
                     const float* __restrict__ wpost2,
                     float* __restrict__ cg, float* __restrict__ qfp, float* __restrict__ b2,
                     _Float16* __restrict__ gw1T, _Float16* __restrict__ A2T)
{
  const int t = threadIdx.x;
  const int cgoff[9] = {CG0,CG1,CG2,CG3,CG4,CG5,CG6,CG7,CG8};
  if (t == 0){
    for (int idx=0; idx<9; ++idx)
      real_cg_dev(INS_I[idx], INS_J[idx], INS_LO[idx], cg + cgoff[idx]);
    // QF = concat_l sqrt(2l+1)*real_cg(l,1,1) reshaped (9,9), then fold PERM
    float QF[81]; float tmp[45];
    int row = 0;
    for (int l=0;l<3;++l){
      real_cg_dev(l,1,1,tmp);
      float sc = sqrtf(2.f*l + 1.f);
      for (int a=0;a<2*l+1;++a)
        for (int c=0;c<9;++c) QF[(row+a)*9+c] = sc * tmp[a*9+c];
      row += 2*l+1;
    }
    const int P[3] = {2,0,1};
    for (int k=0;k<9;++k)
      for (int p=0;p<3;++p)
        for (int q=0;q<3;++q)
          qfp[k*9 + 3*p + q] = QF[k*9 + 3*P[p] + P[q]];
  }
  // gw1 K-transposed to f16: gw1T[n][k] = gw1[k][n]
  for (int e = t; e < 128*64; e += blockDim.x){
    int n = e / 128, k = e % 128;
    gw1T[n*128 + k] = (_Float16)gw1[k*64 + n];
  }
  // folded bias: b2[q] = sum_w c(idx,w)*gb2[idx*256+u*16+w]
  for (int q = t; q < 144; q += blockDim.x){
    int idx = q / 16, u = q % 16;
    const float* wp = (INS_LO[idx]==0) ? wpost0 : (INS_LO[idx]==1 ? wpost1 : wpost2);
    float sc = rsqrtf((float)(INS_NLO[idx]*16)) * 0.25f;   // 0.25 = 1/sqrt(MUL)
    float s = 0.f;
    for (int w=0; w<16; ++w)
      s += wp[INS_POS[idx]*16 + w] * sc * gb2[idx*256 + u*16 + w];
    b2[q] = s;
  }
  // folded + K-transposed A2: A2T[q][h] = sum_w c(idx,w)*gw2[h][idx*256+u*16+w]
  for (int e = t; e < 144*64; e += blockDim.x){
    int q = e / 64, h = e % 64;
    int idx = q / 16, u = q % 16;
    const float* wp = (INS_LO[idx]==0) ? wpost0 : (INS_LO[idx]==1 ? wpost1 : wpost2);
    float sc = rsqrtf((float)(INS_NLO[idx]*16)) * 0.25f;
    float s = 0.f;
    for (int w=0; w<16; ++w)
      s += wp[INS_POS[idx]*16 + w] * sc * gw2[h*2304 + idx*256 + u*16 + w];
    A2T[q*64 + h] = (_Float16)s;
  }
}

// ------------------------------ main kernel --------------------------------

// A operand as loaded here (contiguous k, same as load_b_T): lane(0-15)/lane(16-31) both hold rows M=lane&15;
// VGPR r holds K-pair k,k+1 with k = 2r + 16*laneHi.
__device__ inline v16h load_a_f32(const float* rowBase, int hi){
  v16h a;
#pragma unroll
  for (int r=0;r<8;++r){
    int k = 2*r + 16*hi;
    float2 f = *(const float2*)(rowBase + k);
    a[2*r]   = (_Float16)f.x;
    a[2*r+1] = (_Float16)f.y;
  }
  return a;
}
__device__ inline v16h load_a_h16(const _Float16* rowBase, int hi){
  v16h a;
#pragma unroll
  for (int r=0;r<8;++r){
    int k = 2*r + 16*hi;
    h2v p = *(const h2v*)(rowBase + k);
    a[2*r] = p.x; a[2*r+1] = p.y;
  }
  return a;
}
// B operand as loaded here (32x16, contiguous k, same as A): lanes 0-15 K=0..15, lanes 16-31 K=16..31,
// VGPR r holds K=2r,2r+1 (+16 for hi lanes), column n = lane&15.
// Weights are stored K-transposed so each pair is one aligned 4B load.
__device__ inline v16h load_b_T(const _Float16* colBase){
  v16h b;
#pragma unroll
  for (int r=0;r<8;++r){
    h2v p = *(const h2v*)(colBase + 2*r);
    b[2*r] = p.x; b[2*r+1] = p.y;
  }
  return b;
}

// tensor product inner kernel, component-major packed layout:
// hia[a*8+u2] holds h_i[u=2*u2, a], h_i[u=2*u2+1, a] as a float2.
// y_lo[m] += CG[a,b,m] * sum_u h_i[u,a]*h_j[u,b]*gp[u]
template<int DI,int DJ,int DM>
__device__ inline void tp_accum(const f2* hia, const f2* hjb,
                                const f2* gp2, const float* cgp, float* y){
#pragma unroll
  for (int a=0;a<DI;++a)
#pragma unroll
    for (int b=0;b<DJ;++b){
      f2 acc = {0.f, 0.f};
#pragma unroll
      for (int u2=0;u2<8;++u2) acc += hia[a*8+u2] * hjb[b*8+u2] * gp2[u2];
      float s = acc.x + acc.y;
#pragma unroll
      for (int m=0;m<DM;++m) y[m] += cgp[(a*DJ+b)*DM+m] * s;
    }
}

__device__ inline void load_gp(const float* GPl, int idx, f2* gp2){
  const f2* gl = (const f2*)(GPl + idx*16);
#pragma unroll
  for (int u2=0;u2<8;++u2) gp2[u2] = gl[u2];
}

__global__ __launch_bounds__(64)
void csc_main_kernel(const float* __restrict__ xsc, const float* __restrict__ xsp,
                     const float* __restrict__ wp0, const float* __restrict__ wp1,
                     const float* __restrict__ wp2, const float* __restrict__ gb1,
                     const float* __restrict__ cg,  const float* __restrict__ qfp,
                     const float* __restrict__ b2,
                     const _Float16* __restrict__ gw1T, const _Float16* __restrict__ A2T,
                     float* __restrict__ out)
{
  __shared__ _Float16 Slds[2][32*64];    // silu(x@gw1+b) per wave, f16
  __shared__ float    GPlds[2][32*144];  // folded generator output per wave
  const int wave = threadIdx.x >> 5;
  const int lane = threadIdx.x & 31;
  const int n15  = lane & 15;
  const int hi   = lane >> 4;
  const int atomBase = (blockIdx.x*2 + wave) * 32;
  _Float16* S  = Slds[wave];
  float*    GP = GPlds[wave];

  // prefetch this lane's x_spherical row (1920B) so the tail phase hits cache;
  // overlaps HBM latency with the two WMMA GEMM phases below.
  const int    atom = atomBase + lane;
  const float* xrow = xsp + (size_t)atom * 480;
#pragma unroll
  for (int i=0;i<15;++i)
    __builtin_prefetch((const char*)xrow + i*128, 0, 3);

  // ---------------- GEMM1: S = silu(Xs @ gw1 + gb1), 32x128 @ 128x64 -------
  v16h A1[2][4];
#pragma unroll
  for (int t=0;t<2;++t)
#pragma unroll
    for (int c=0;c<4;++c)
      A1[t][c] = load_a_f32(xsc + (size_t)(atomBase + t*16 + n15)*128 + c*32, hi);

#pragma unroll
  for (int j=0;j<4;++j){
    v8f acc0, acc1;
#pragma unroll
    for (int r=0;r<8;++r){ acc0[r] = 0.f; acc1[r] = 0.f; }
#pragma unroll
    for (int c=0;c<4;++c){
      v16h B = load_b_T(gw1T + (j*16 + n15)*128 + c*32 + 16*hi);
      acc0 = __builtin_amdgcn_wmma_f32_16x16x32_f16(false, A1[0][c], false, B, (short)0, acc0, false, false);
      acc1 = __builtin_amdgcn_wmma_f32_16x16x32_f16(false, A1[1][c], false, B, (short)0, acc1, false, false);
    }
    float bias = gb1[j*16 + n15];
#pragma unroll
    for (int r=0;r<8;++r){
      float v0 = acc0[r] + bias;
      float v1 = acc1[r] + bias;
      v0 = v0 / (1.f + __expf(-v0));        // silu
      v1 = v1 / (1.f + __expf(-v1));
      S[(hi*8 + r)*64      + j*16 + n15] = (_Float16)v0;
      S[(16 + hi*8 + r)*64 + j*16 + n15] = (_Float16)v1;
    }
  }
  asm volatile("s_wait_dscnt 0" ::: "memory");

  // ---------------- GEMM2: GP = S @ A2 + b2, 32x64 @ 64x144 ----------------
  v16h A2op[2][2];
#pragma unroll
  for (int t=0;t<2;++t)
#pragma unroll
    for (int c=0;c<2;++c)
      A2op[t][c] = load_a_h16(S + (t*16 + n15)*64 + c*32, hi);

#pragma unroll
  for (int j=0;j<9;++j){
    float bv = b2[j*16 + n15];
    v8f acc0, acc1;
#pragma unroll
    for (int r=0;r<8;++r){ acc0[r] = bv; acc1[r] = bv; }
#pragma unroll
    for (int c=0;c<2;++c){
      v16h B = load_b_T(A2T + (j*16 + n15)*64 + c*32 + 16*hi);
      acc0 = __builtin_amdgcn_wmma_f32_16x16x32_f16(false, A2op[0][c], false, B, (short)0, acc0, false, false);
      acc1 = __builtin_amdgcn_wmma_f32_16x16x32_f16(false, A2op[1][c], false, B, (short)0, acc1, false, false);
    }
#pragma unroll
    for (int r=0;r<8;++r){
      GP[(hi*8 + r)*144      + j*16 + n15] = acc0[r];
      GP[(16 + hi*8 + r)*144 + j*16 + n15] = acc1[r];
    }
  }
  asm volatile("s_wait_dscnt 0" ::: "memory");

  // ---------------- per-lane phase: one atom per lane ----------------------
  // component-major packed layout: hv[comp][u2] = {h[2*u2,comp], h[2*u2+1,comp]}
  const float* GPl = GP + lane*144;

  // h0: hv0[u2] = sum_c x0[c]*wp0[c][u-pair] / sqrt(128)
  f2 hv0[8] = {};
  {
    const f2* wv = (const f2*)wp0;
    for (int c=0;c<128;++c){
      float xv = xrow[c];
      const f2* wr = wv + c*8;
#pragma unroll
      for (int u2=0;u2<8;++u2) hv0[u2] += wr[u2] * xv;
    }
    const float s0 = 0.0883883476483184f; // 1/sqrt(128)
#pragma unroll
    for (int u2=0;u2<8;++u2) hv0[u2] *= s0;
  }
  // h1: hv1[m*8+u2] = sum_c x1[c][m]*wp1[c][u-pair] / 8
  f2 hv1[24] = {};
  {
    const float* x1 = xrow + 128;
    const f2*    wv = (const f2*)wp1;
    for (int c=0;c<64;++c){
      float xa = x1[c*3+0], xb = x1[c*3+1], xc = x1[c*3+2];
      const f2* wr = wv + c*8;
#pragma unroll
      for (int u2=0;u2<8;++u2){
        f2 w = wr[u2];
        hv1[0*8+u2] += w * xa;
        hv1[1*8+u2] += w * xb;
        hv1[2*8+u2] += w * xc;
      }
    }
#pragma unroll
    for (int e=0;e<24;++e) hv1[e] *= 0.125f;
  }
  // h2: hv2[m*8+u2] = sum_c x2[c][m]*wp2[c][u-pair] / sqrt(32)
  f2 hv2[40] = {};
  {
    const float* x2 = xrow + 320;
    const f2*    wv = (const f2*)wp2;
    for (int c=0;c<32;++c){
      float xm[5];
#pragma unroll
      for (int m=0;m<5;++m) xm[m] = x2[c*5+m];
      const f2* wr = wv + c*8;
#pragma unroll
      for (int u2=0;u2<8;++u2){
        f2 w = wr[u2];
#pragma unroll
        for (int m=0;m<5;++m) hv2[m*8+u2] += w * xm[m];
      }
    }
    const float s2 = 0.1767766952966369f; // 1/sqrt(32)
#pragma unroll
    for (int e=0;e<40;++e) hv2[e] *= s2;
  }

  // tensor product: y_lo[m] += sum_{a,b} CG[a,b,m] * sum_u hi[u,a]hj[u,b]gp[u]
  float y0[1] = {}; float y1[3] = {}; float y2[5] = {};
  f2 gp2[8];
  load_gp(GPl, 0, gp2); tp_accum<1,1,1>(hv0, hv0, gp2, cg + CG0, y0);
  load_gp(GPl, 1, gp2); tp_accum<1,5,5>(hv0, hv2, gp2, cg + CG1, y2);
  load_gp(GPl, 2, gp2); tp_accum<3,3,1>(hv1, hv1, gp2, cg + CG2, y0);
  load_gp(GPl, 3, gp2); tp_accum<3,3,3>(hv1, hv1, gp2, cg + CG3, y1);
  load_gp(GPl, 4, gp2); tp_accum<3,3,5>(hv1, hv1, gp2, cg + CG4, y2);
  load_gp(GPl, 5, gp2); tp_accum<5,1,5>(hv2, hv0, gp2, cg + CG5, y2);
  load_gp(GPl, 6, gp2); tp_accum<5,5,1>(hv2, hv2, gp2, cg + CG6, y0);
  load_gp(GPl, 7, gp2); tp_accum<5,5,3>(hv2, hv2, gp2, cg + CG7, y1);
  load_gp(GPl, 8, gp2); tp_accum<5,5,5>(hv2, hv2, gp2, cg + CG8, y2);

  // out[p][q] = sum_k ao[k] * QFp[k][3p+q]   (PERM folded into QFp)
  float ao[9] = {y0[0], y1[0],y1[1],y1[2], y2[0],y2[1],y2[2],y2[3],y2[4]};
  float* orow = out + (size_t)atom * 9;
#pragma unroll
  for (int e=0;e<9;++e){
    float s = 0.f;
#pragma unroll
    for (int k=0;k<9;++k) s += ao[k] * qfp[k*9 + e];
    orow[e] = s;
  }
}

// ------------------------------- launch ------------------------------------

extern "C" void kernel_launch(void* const* d_in, const int* in_sizes, int n_in,
                              void* d_out, int out_size, void* d_ws, size_t ws_size,
                              hipStream_t stream)
{
  const float* xsc    = (const float*)d_in[0];
  const float* xsp    = (const float*)d_in[1];
  const float* wp0    = (const float*)d_in[2];
  const float* wp1    = (const float*)d_in[3];
  const float* wp2    = (const float*)d_in[4];
  const float* gw1    = (const float*)d_in[5];
  const float* gb1    = (const float*)d_in[6];
  const float* gw2    = (const float*)d_in[7];
  const float* gb2    = (const float*)d_in[8];
  const float* wpost0 = (const float*)d_in[9];
  const float* wpost1 = (const float*)d_in[10];
  const float* wpost2 = (const float*)d_in[11];

  char* ws = (char*)d_ws;
  float*    cg   = (float*)(ws + WS_CG);
  float*    qfp  = (float*)(ws + WS_QFP);
  float*    b2   = (float*)(ws + WS_B2);
  _Float16* gw1T = (_Float16*)(ws + WS_GW1T);
  _Float16* A2T  = (_Float16*)(ws + WS_A2T);

  csc_prep_kernel<<<1, 256, 0, stream>>>(gw1, gw2, gb2, wpost0, wpost1, wpost2,
                                         cg, qfp, b2, gw1T, A2T);
  csc_main_kernel<<<dim3(N_ATOM/64), 64, 0, stream>>>(xsc, xsp, wp0, wp1, wp2, gb1,
                                                      cg, qfp, b2, gw1T, A2T,
                                                      (float*)d_out);
}
